// SparseMHAEncoder_3582002725458
// MI455X (gfx1250) — matrix operand order repaired and checked by reading, not compiled, not run on hardware
//
#include <hip/hip_runtime.h>
#include <hip/hip_bf16.h>

// ---------------------------------------------------------------------------
// Sliding-window MHA encoder for MI455X (gfx1250, wave32, WMMA 16x16x32 f16)
//
//   WqT/WkT/WvT/WoT = f16 transposed weights (N x K)          [wt_convert]
//   Q = q@Wq, K = k@Wk (f16, (b,t,h,d))                       [gemm_wmma]
//   V = v@Wv stored transposed+padded (b,h,d, 64|t|32 pad)    [gemm_wmma<VSTORE>]
//   banded scores -> softmax -> P.V with probs staged in LDS  [attn_window_wmma]
//   out = O@Wout (f32)                                        [gemm_wmma]
//
// GEMM: 32x64 C macro-tile per wave (8 WMMAs / 12 b128 loads per k-step).
// All global accesses use uniform-base + u32 per-lane offsets so the backend
// emits SADDR-form global_load/store (one voffset VGPR + immediate per
// stream) instead of per-lane 64-bit pointer arithmetic.
// ---------------------------------------------------------------------------

typedef __attribute__((ext_vector_type(16))) _Float16 v16h;
typedef __attribute__((ext_vector_type(8)))  _Float16 v8h;
typedef __attribute__((ext_vector_type(8)))  float    v8f;

static __device__ __forceinline__ v8f wmma_f16(v16h a, v16h b, v8f c) {
  return __builtin_amdgcn_wmma_f32_16x16x32_f16(false, a, false, b,
                                                (short)0, c, false, false);
}
static __device__ __forceinline__ v16h cat8(v8h lo, v8h hi) {
  return __builtin_shufflevector(lo, hi, 0, 1, 2, 3, 4, 5, 6, 7,
                                 8, 9, 10, 11, 12, 13, 14, 15);
}
// uniform base + u32 element offset -> saddr-form b128 loads
static __device__ __forceinline__ v8h load8(const _Float16* base, unsigned off) {
  return *(const v8h*)(base + off);
}
static __device__ __forceinline__ v8h load8(const float* base, unsigned off) {
  v8f x = *(const v8f*)(base + off);           // 2x global_load_b128 (saddr)
  return __builtin_convertvector(x, v8h);      // v_cvt_pk_f16_f32 x4
}
static __device__ __forceinline__ v8h load8lds(const _Float16* p) {
  return *(const v8h*)p;                       // ds_load_b128
}

constexpr int Bsz  = 2;
constexpr int LQ   = 2048;
constexpr int LKV  = 2048;
constexpr int Hn   = 8;
constexpr int DQK  = 64;
constexpr int DV   = 64;
constexpr int DIM  = 512;                 // all model dims
constexpr int SPAN = 64;
constexpr int VTP  = 64 + LKV + 32;       // padded V^T row: [64 pad | t | 32 pad]

// ---------------------------------------------------------------------------
// Weight convert+transpose: Wt[n*K + k] = (f16) W[k*N + n]
// ---------------------------------------------------------------------------
__global__ __launch_bounds__(256)
void wt_convert(const float* __restrict__ W, _Float16* __restrict__ Wt,
                int K, int N) {
  const unsigned i = blockIdx.x * blockDim.x + threadIdx.x;
  if (i >= (unsigned)(K * N)) return;
  const unsigned kk = i / N, n = i - kk * N;
  Wt[n * (unsigned)K + kk] = (_Float16)W[i];
}

// Zero the front/tail pads of every V^T row so pad values are exactly 0.
__global__ __launch_bounds__(256)
void zero_v_pads(_Float16* __restrict__ Vt) {
  const unsigned i = blockIdx.x * blockDim.x + threadIdx.x;  // rows * 96
  const unsigned row = i / 96, j = i - row * 96;
  const unsigned off = (j < 64) ? j : (64 + LKV + (j - 64));
  Vt[row * (unsigned)VTP + off] = (_Float16)0.f;
}

// ---------------------------------------------------------------------------
// 32x64-macro-tile WMMA GEMM: C[M,N] = A[M,K] * Bt^T  (Bt f16, N x K row-major)
// One wave per 32(M) x 64(N) tile; K consumed 32/step.
// VSTORE: scatter D into the padded V^T (b,h,d,t) layout.
// ---------------------------------------------------------------------------
template <typename TA, typename TO, bool VSTORE>
__global__ __launch_bounds__(128)
void gemm_wmma(const TA* __restrict__ A, const _Float16* __restrict__ Bt,
               TO* __restrict__ C, int M, int N, int K) {
  const int lane = threadIdx.x & 31;
  const int wave = threadIdx.x >> 5;
  const int tile = blockIdx.x * (blockDim.x >> 5) + wave;
  const int ntn  = N >> 6;           // 64-wide N macro-tiles
  const int tM   = tile / ntn;       // 32-tall M macro-tiles
  const int tN   = tile % ntn;
  if (tM * 32 >= M) return;

  const int  lq  = lane & 15;
  const bool hi  = lane >= 16;
  const unsigned aK0 = hi ? 16u : 0u;    // A elems 0..7  -> K = aK0+e
  const unsigned aK1 = hi ? 24u : 8u;    // A elems 8..15 -> K = aK1+(e-8)
  const unsigned bK0 = hi ? 16u : 0u;    // B elem  e     -> K = bK0+e

  // u32 element offsets (loop-invariant per stream) -> saddr-form loads
  const unsigned a0 = (unsigned)(tM * 32 + lq) * (unsigned)K;
  const unsigned a1 = a0 + 16u * (unsigned)K;
  const unsigned b0 = (unsigned)(tN * 64 + lq) * (unsigned)K;
  const unsigned b1 = b0 + 16u * (unsigned)K;
  const unsigned b2 = b0 + 32u * (unsigned)K;
  const unsigned b3 = b0 + 48u * (unsigned)K;

  v8f acc[2][4] = {};
  for (unsigned kk = 0; kk < (unsigned)K; kk += 32) {
    __builtin_prefetch(A + a0 + kk + 32, 0, 0);          // global_prefetch_b8
    __builtin_prefetch(Bt + b0 + kk + 32, 0, 0);
    const v16h fa0 = cat8(load8(A, a0 + kk + aK0), load8(A, a0 + kk + aK1));
    const v16h fa1 = cat8(load8(A, a1 + kk + aK0), load8(A, a1 + kk + aK1));
    const v16h fb0 = cat8(load8(Bt, b0 + kk + bK0), load8(Bt, b0 + kk + bK0 + 8));
    acc[0][0] = wmma_f16(fa0, fb0, acc[0][0]);
    acc[1][0] = wmma_f16(fa1, fb0, acc[1][0]);
    const v16h fb1 = cat8(load8(Bt, b1 + kk + bK0), load8(Bt, b1 + kk + bK0 + 8));
    acc[0][1] = wmma_f16(fa0, fb1, acc[0][1]);
    acc[1][1] = wmma_f16(fa1, fb1, acc[1][1]);
    const v16h fb2 = cat8(load8(Bt, b2 + kk + bK0), load8(Bt, b2 + kk + bK0 + 8));
    acc[0][2] = wmma_f16(fa0, fb2, acc[0][2]);
    acc[1][2] = wmma_f16(fa1, fb2, acc[1][2]);
    const v16h fb3 = cat8(load8(Bt, b3 + kk + bK0), load8(Bt, b3 + kk + bK0 + 8));
    acc[0][3] = wmma_f16(fa0, fb3, acc[0][3]);
    acc[1][3] = wmma_f16(fa1, fb3, acc[1][3]);
  }

  // D-layout: VGPR r -> M = r (+8 for lanes 16..31), N = lane&15
#pragma unroll
  for (int i = 0; i < 2; ++i) {
    const int orow = tM * 32 + i * 16 + (hi ? 8 : 0);
#pragma unroll
    for (int j = 0; j < 4; ++j) {
      const int col = tN * 64 + j * 16 + lq;
      if constexpr (VSTORE) {
        // scatter into padded V^T: Vt[((b*H+h)*DV+d)*VTP + 64 + t]
        const int hh = col >> 6, d = col & 63;
#pragma unroll
        for (int r = 0; r < 8; ++r) {
          const int gt = orow + r;
          const int bb = gt / LQ, t = gt - bb * LQ;
          const unsigned o =
              (unsigned)((bb * Hn + hh) * DV + d) * (unsigned)VTP + 64u + (unsigned)t;
          C[o] = (TO)acc[i][j][r];
        }
      } else {
#pragma unroll
        for (int r = 0; r < 8; ++r)
          C[(unsigned)(orow + r) * (unsigned)N + (unsigned)col] = (TO)acc[i][j][r];
      }
    }
  }
}

// ---------------------------------------------------------------------------
// Banded attention: one wave per (b, h, 16 queries t0..t0+15).
// Keys span [t0-64, t0+15] -> 5 key tiles (80 keys, padded to 96 for P.V).
// ---------------------------------------------------------------------------
__global__ __launch_bounds__(128)
void attn_window_wmma(const _Float16* __restrict__ Qh,
                      const _Float16* __restrict__ Kh,
                      const _Float16* __restrict__ Vt,
                      _Float16* __restrict__ Oh) {
  constexpr int WAVES = 4;
  __shared__ __align__(16) _Float16 Pl[WAVES][16 * 96];  // P^T: [q][s=0..95]

  const int lane = threadIdx.x & 31;
  const int wave = threadIdx.x >> 5;
  const int wid  = blockIdx.x * WAVES + wave;
  const int nqb  = LQ / 16;
  const int qb   = wid % nqb;
  const int h    = (wid / nqb) % Hn;
  const int b    = wid / (nqb * Hn);
  const int t0   = qb * 16;

  const int  lq  = lane & 15;
  const bool hi  = lane >= 16;
  const unsigned aK0 = hi ? 16u : 0u;
  const unsigned aK1 = hi ? 24u : 8u;
  const unsigned bK0 = hi ? 16u : 0u;
  const float scale = 0.125f;                 // 1/sqrt(DQK)

  // ---- hoist Q fragments (B operand of scores), reused by all 5 key tiles -
  const unsigned qoff = (unsigned)((b * LQ + t0 + lq) * Hn + h) * DQK;
  v16h qq[2];
#pragma unroll
  for (int kst = 0; kst < 2; ++kst)
    qq[kst] = cat8(load8(Qh, qoff + kst * 32 + bK0),
                   load8(Qh, qoff + kst * 32 + bK0 + 8));

  // ---- scores: D[key, query] = K[key] . Q[query] --------------------------
  v8f sacc[5];
#pragma unroll
  for (int blk = 0; blk < 5; ++blk) {
    const int key  = t0 - 64 + blk * 16 + lq;   // A row = key (per lane)
    const int keyc = key < 0 ? 0 : key;         // clamp; masked later
    const unsigned koff = (unsigned)((b * LKV + keyc) * Hn + h) * DQK;
    v8f acc = {};
#pragma unroll
    for (int kst = 0; kst < 2; ++kst) {         // DQK = 64 = 2 x 32
      v16h a = cat8(load8(Kh, koff + kst * 32 + aK0),
                    load8(Kh, koff + kst * 32 + aK1));
      acc = wmma_f16(a, qq[kst], acc);
    }
    sacc[blk] = acc;
  }

  // ---- mask + softmax over 80 keys (query column split lane <-> lane^16) --
  const int t  = t0 + lq;
  const int lo = (t - (SPAN - 1)) < 0 ? 0 : (t - (SPAN - 1));
  float m = -__builtin_inff();
#pragma unroll
  for (int blk = 0; blk < 5; ++blk) {
#pragma unroll
    for (int r = 0; r < 8; ++r) {
      const int key = t0 - 64 + blk * 16 + r + (hi ? 8 : 0);
      const bool valid = (key >= lo) && (key <= t);
      const float s = valid ? sacc[blk][r] * scale : -__builtin_inff();
      sacc[blk][r] = s;
      m = fmaxf(m, s);
    }
  }
  m = fmaxf(m, __shfl_xor(m, 16, 32));
  float sum = 0.f;
#pragma unroll
  for (int blk = 0; blk < 5; ++blk) {
#pragma unroll
    for (int r = 0; r < 8; ++r) {
      const float e = __expf(sacc[blk][r] - m);
      sacc[blk][r] = e;
      sum += e;
    }
  }
  sum += __shfl_xor(sum, 16, 32);
  const float inv = 1.f / sum;                  // key==t always valid -> >0

  // ---- stage P^T (f16) in LDS: Pt[q*96 + s], zero-pad s in [80,96) --------
  _Float16* Pt = Pl[wave];
#pragma unroll
  for (int blk = 0; blk < 5; ++blk) {
#pragma unroll
    for (int r = 0; r < 8; ++r) {
      const int s = blk * 16 + r + (hi ? 8 : 0);
      Pt[lq * 96 + s] = (_Float16)(sacc[blk][r] * inv);
    }
  }
  {
    const int q = lane >> 1, s0 = 80 + (lane & 1) * 8;
#pragma unroll
    for (int i = 0; i < 8; ++i) Pt[q * 96 + s0 + i] = (_Float16)0.f;
  }
  __syncthreads();

  // ---- O[query, d] = P^T(16x96) x V(96x64): contiguous b128 everywhere ----
  const unsigned vhead = (unsigned)(b * Hn + h) * DV;
  v8f oacc[4] = {};
#pragma unroll
  for (int kst = 0; kst < 3; ++kst) {
    const _Float16* pr = Pt + lq * 96 + kst * 32;   // A = P^T row (query lq)
    v16h a = cat8(load8lds(pr + aK0), load8lds(pr + aK1));
#pragma unroll
    for (int nt = 0; nt < 4; ++nt) {
      // B elem e -> key = t0-64 + kst*32 + bK0 + e ; +64 pad cancels the -64
      const unsigned voff =
          (vhead + nt * 16 + lq) * (unsigned)VTP + t0 + kst * 32 + bK0;
      v16h vv = cat8(load8(Vt, voff), load8(Vt, voff + 8));
      oacc[nt] = wmma_f16(a, vv, oacc[nt]);
    }
  }

  // ---- store O (f16, row-major (b*LQ+t) x DIM) ----------------------------
  const int orow = t0 + (hi ? 8 : 0);
#pragma unroll
  for (int nt = 0; nt < 4; ++nt) {
#pragma unroll
    for (int r = 0; r < 8; ++r) {
      const unsigned o =
          (unsigned)((b * LQ + orow + r) * Hn + h) * DV + nt * 16 + lq;
      Oh[o] = (_Float16)oacc[nt][r];
    }
  }
}

// ---------------------------------------------------------------------------
extern "C" void kernel_launch(void* const* d_in, const int* in_sizes, int n_in,
                              void* d_out, int out_size, void* d_ws, size_t ws_size,
                              hipStream_t stream) {
  const float* q  = (const float*)d_in[0];
  const float* k  = (const float*)d_in[1];
  const float* v  = (const float*)d_in[2];
  const float* Wq = (const float*)d_in[3];
  const float* Wk = (const float*)d_in[4];
  const float* Wv = (const float*)d_in[5];
  const float* Wo = (const float*)d_in[6];
  float* out = (float*)d_out;

  const int M = Bsz * LQ;   // 4096
  const int N = DIM;        // 512
  const int K = DIM;        // 512

  // f16 workspace layout (elements):
  //   4 transposed weights (256K each), Qh/Kh (2M each), Vt (padded), Oh
  _Float16* WqT = (_Float16*)d_ws;
  _Float16* WkT = WqT + (size_t)DIM * DIM;
  _Float16* WvT = WkT + (size_t)DIM * DIM;
  _Float16* WoT = WvT + (size_t)DIM * DIM;
  _Float16* Qh  = WoT + (size_t)DIM * DIM;
  _Float16* Kh  = Qh + (size_t)M * N;
  _Float16* Vt  = Kh + (size_t)M * N;                      // B*H*DV x VTP
  _Float16* Oh  = Vt + (size_t)Bsz * Hn * DV * VTP;
  // total ~9.54M f16 = ~18.2 MiB

  // weights: convert + transpose to f16
  const int wcount = DIM * DIM;
  dim3 wblk(256), wgrd((wcount + 255) / 256);
  wt_convert<<<wgrd, wblk, 0, stream>>>(Wq, WqT, K, N);
  wt_convert<<<wgrd, wblk, 0, stream>>>(Wk, WkT, K, N);
  wt_convert<<<wgrd, wblk, 0, stream>>>(Wv, WvT, K, N);
  wt_convert<<<wgrd, wblk, 0, stream>>>(Wo, WoT, K, N);

  // zero the V^T pads (96 per row, B*H*DV rows = 98304 elements)
  zero_v_pads<<<dim3(Bsz * Hn * DV * 96 / 256), dim3(256), 0, stream>>>(Vt);

  const int tiles = (M / 32) * (N / 64);   // 1024 macro-tiles, 4 waves/block
  dim3 gblk(128), ggrd(tiles / 4);
  gemm_wmma<float, _Float16, false><<<ggrd, gblk, 0, stream>>>(q, WqT, Qh, M, N, K);
  gemm_wmma<float, _Float16, false><<<ggrd, gblk, 0, stream>>>(k, WkT, Kh, M, N, K);
  gemm_wmma<float, _Float16, true ><<<ggrd, gblk, 0, stream>>>(v, WvT, Vt, M, N, K);

  const int awaves = Bsz * Hn * (LQ / 16);  // 2048 query tiles
  attn_window_wmma<<<dim3(awaves / 4), dim3(128), 0, stream>>>(Qh, Kh, Vt, Oh);

  gemm_wmma<_Float16, float, false><<<ggrd, gblk, 0, stream>>>(Oh, WoT, out, M, N, K);
}
